// CLSAggregator_4578435138003
// MI455X (gfx1250) — compile-verified, matrix operand order checked by reading, not run on hardware
//
#include <hip/hip_runtime.h>
#include <hip/hip_bf16.h>

// ---------------- model dims ----------------
static constexpr int Bb   = 4;
static constexpr int SEQ  = 1024;   // N=1023 tokens + 1 CLS
static constexpr int Dm   = 768;
static constexpr int Hh   = 16;
static constexpr int HD   = 48;
static constexpr int FF   = 3072;
static constexpr int LAY  = 6;
static constexpr int Mrow = Bb * SEQ;   // 4096

typedef __bf16 bf16;
typedef __attribute__((ext_vector_type(16))) __bf16 v16bf;
typedef __attribute__((ext_vector_type(8)))  __bf16 v8bf;
typedef __attribute__((ext_vector_type(8)))  float  v8f;

__device__ __forceinline__ v8f wmma_bf(v16bf a, v16bf b, v8f c) {
  return __builtin_amdgcn_wmma_f32_16x16x32_bf16(false, a, false, b, (short)0, c, false, false);
}

// CDNA5 async global->LDS copy (tracked by ASYNCcnt), 16B per lane.
__device__ __forceinline__ void async_cp_b128(unsigned lds_off, const void* gptr) {
  asm volatile("global_load_async_to_lds_b128 %0, %1, off"
               :: "v"(lds_off), "v"(gptr) : "memory");
}
__device__ __forceinline__ void wait_async0() {
  asm volatile("s_wait_asynccnt 0" ::: "memory");
}

// Load one 16x32 bf16 operand fragment (A- or B-style, K-major rows).
// Lane l: row = (l&15) chosen by caller via rowbase; hi = l>>4 selects the
// K-split per ISA: chunks at k = 8*hi and k = 16 + 8*hi.
__device__ __forceinline__ v16bf ld_frag(const bf16* rowbase, int hi) {
  v8bf lo = *(const v8bf*)(rowbase + 8 * hi);
  v8bf hh = *(const v8bf*)(rowbase + 16 + 8 * hi);
  v16bf r;
#pragma unroll
  for (int i = 0; i < 8; ++i) { r[i] = lo[i]; r[8 + i] = hh[i]; }
  return r;
}

// ---------------- elementwise kernels ----------------
__global__ __launch_bounds__(256) void prep_x(const float* __restrict__ toks,
                                              const float* __restrict__ cls,
                                              float* __restrict__ x) {
  size_t i = (size_t)blockIdx.x * 256 + threadIdx.x;   // over Bb*SEQ*Dm
  int d = (int)(i % Dm);
  size_t row = i / Dm;
  int t = (int)(row % SEQ);
  size_t b = row / SEQ;
  x[i] = (t == 0) ? cls[d] : toks[((b * (SEQ - 1)) + (t - 1)) * (size_t)Dm + d];
}

__global__ __launch_bounds__(256) void f32_to_bf16(const float* __restrict__ src,
                                                   bf16* __restrict__ dst, size_t n) {
  size_t i = (size_t)blockIdx.x * 256 + threadIdx.x;
  if (i < n) dst[i] = (bf16)src[i];
}

__global__ __launch_bounds__(256) void silu_mul(const float* __restrict__ g,
                                                const float* __restrict__ u,
                                                bf16* __restrict__ out) {
  size_t i = (size_t)blockIdx.x * 256 + threadIdx.x;   // Mrow*FF, exact multiple
  float gv = g[i];
  float s = gv / (1.0f + __expf(-gv));
  out[i] = (bf16)(s * u[i]);
}

// LayerNorm of one 768-elem row -> bf16
__global__ __launch_bounds__(256) void ln_to_bf16(const float* __restrict__ x,
                                                  const float* __restrict__ w,
                                                  const float* __restrict__ bi,
                                                  bf16* __restrict__ out) {
  const int row = blockIdx.x;
  const float* xr = x + (size_t)row * Dm;
  const int tid = threadIdx.x, lane = tid & 31, wv = tid >> 5;
  float s = 0.f, s2 = 0.f;
  for (int i = tid; i < Dm; i += 256) { float v = xr[i]; s += v; s2 += v * v; }
#pragma unroll
  for (int m = 1; m < 32; m <<= 1) { s += __shfl_xor(s, m, 32); s2 += __shfl_xor(s2, m, 32); }
  __shared__ float w1[8], w2[8];
  if (lane == 0) { w1[wv] = s; w2[wv] = s2; }
  __syncthreads();
  float t1 = 0.f, t2 = 0.f;
#pragma unroll
  for (int k = 0; k < 8; ++k) { t1 += w1[k]; t2 += w2[k]; }
  float mean = t1 * (1.0f / Dm);
  float var  = t2 * (1.0f / Dm) - mean * mean;
  float rstd = rsqrtf(var + 1e-5f);
  for (int i = tid; i < Dm; i += 256)
    out[(size_t)row * Dm + i] = (bf16)((xr[i] - mean) * rstd * w[i] + bi[i]);
}

// Final LayerNorm of x[:,0,:] -> f32 out [Bb, Dm]
__global__ __launch_bounds__(256) void ln_final(const float* __restrict__ x,
                                                const float* __restrict__ w,
                                                const float* __restrict__ bi,
                                                float* __restrict__ out) {
  const int b = blockIdx.x;
  const float* xr = x + (size_t)b * SEQ * Dm;
  const int tid = threadIdx.x, lane = tid & 31, wv = tid >> 5;
  float s = 0.f, s2 = 0.f;
  for (int i = tid; i < Dm; i += 256) { float v = xr[i]; s += v; s2 += v * v; }
#pragma unroll
  for (int m = 1; m < 32; m <<= 1) { s += __shfl_xor(s, m, 32); s2 += __shfl_xor(s2, m, 32); }
  __shared__ float w1[8], w2[8];
  if (lane == 0) { w1[wv] = s; w2[wv] = s2; }
  __syncthreads();
  float t1 = 0.f, t2 = 0.f;
#pragma unroll
  for (int k = 0; k < 8; ++k) { t1 += w1[k]; t2 += w2[k]; }
  float mean = t1 * (1.0f / Dm);
  float rstd = rsqrtf(t2 * (1.0f / Dm) - mean * mean + 1e-5f);
  for (int i = tid; i < Dm; i += 256)
    out[(size_t)b * Dm + i] = (xr[i] - mean) * rstd * w[i] + bi[i];
}

// ---------------- WMMA GEMM: C[M,N] = A[M,K](bf16) * W[N,K](bf16)^T (+res) ----
// block = 256 threads (8 waves), tile 128x128, K step 32, double-buffered LDS
// filled by CDNA5 async global->LDS copies (ASYNCcnt).
static constexpr int LSTR = 40;  // LDS row stride (bf16 elems), 80B: 16B aligned

__device__ __forceinline__ void stage_tile(const bf16* __restrict__ A,
                                           const bf16* __restrict__ W,
                                           bf16* As, bf16* Bs,
                                           int bm, int bn, int K, int k0, int tid) {
#pragma unroll
  for (int i = 0; i < 2; ++i) {
    int chunk = tid + 256 * i;           // 0..511
    int r = chunk >> 2, c = (chunk & 3) * 8;
    async_cp_b128((unsigned)(uintptr_t)&As[r * LSTR + c],
                  &A[(size_t)(bm + r) * K + k0 + c]);
    async_cp_b128((unsigned)(uintptr_t)&Bs[r * LSTR + c],
                  &W[(size_t)(bn + r) * K + k0 + c]);
  }
}

template <bool HasRes>
__global__ __launch_bounds__(256)
void gemm_bf16_nt(const bf16* __restrict__ A, const bf16* __restrict__ W,
                  const float* __restrict__ res, float* __restrict__ C,
                  int M, int N, int K) {
  __shared__ bf16 As[2][128 * LSTR];
  __shared__ bf16 Bs[2][128 * LSTR];
  const int tid = threadIdx.x, lane = tid & 31, wave = tid >> 5;
  const int ln = lane & 15, hi = lane >> 4;
  const int bm = blockIdx.y * 128, bn = blockIdx.x * 128;
  const int wm = (wave >> 2) * 64;   // 0,64
  const int wn = (wave & 3) * 32;    // 0,32,64,96

  v8f acc[4][2];
#pragma unroll
  for (int i = 0; i < 4; ++i)
#pragma unroll
    for (int j = 0; j < 2; ++j) acc[i][j] = (v8f)(0.f);

  const int nk = K >> 5;
  stage_tile(A, W, As[0], Bs[0], bm, bn, K, 0, tid);
  wait_async0();
  __syncthreads();

  for (int kt = 0; kt < nk; ++kt) {
    const bf16* Ac = As[kt & 1];
    const bf16* Bc = Bs[kt & 1];
    if (kt + 1 < nk)   // prefetch next tile while computing this one
      stage_tile(A, W, As[(kt + 1) & 1], Bs[(kt + 1) & 1], bm, bn, K, (kt + 1) * 32, tid);

    v16bf af[4], bfr[2];
#pragma unroll
    for (int mt = 0; mt < 4; ++mt) af[mt] = ld_frag(&Ac[(wm + 16 * mt + ln) * LSTR], hi);
#pragma unroll
    for (int nt = 0; nt < 2; ++nt) bfr[nt] = ld_frag(&Bc[(wn + 16 * nt + ln) * LSTR], hi);
#pragma unroll
    for (int mt = 0; mt < 4; ++mt)
#pragma unroll
      for (int nt = 0; nt < 2; ++nt) acc[mt][nt] = wmma_bf(af[mt], bfr[nt], acc[mt][nt]);

    wait_async0();
    __syncthreads();
  }

#pragma unroll
  for (int mt = 0; mt < 4; ++mt)
#pragma unroll
    for (int nt = 0; nt < 2; ++nt) {
      int n = bn + wn + nt * 16 + ln;
      int mb = bm + wm + mt * 16 + hi * 8;
      if (HasRes) {
        float rv[8];
#pragma unroll
        for (int r = 0; r < 8; ++r) rv[r] = res[(size_t)(mb + r) * N + n];
#pragma unroll
        for (int r = 0; r < 8; ++r) C[(size_t)(mb + r) * N + n] = acc[mt][nt][r] + rv[r];
      } else {
#pragma unroll
        for (int r = 0; r < 8; ++r) C[(size_t)(mb + r) * N + n] = acc[mt][nt][r];
      }
    }
}

// ---------------- flash attention with ALiBi ----------------
// grid.x = Bb*Hh*(SEQ/128); wave w owns 16 q-rows; all 128 k-cols per block.
__global__ __launch_bounds__(256)
void attn_flash(const float* __restrict__ qkv, const float* __restrict__ log_slopes,
                bf16* __restrict__ obuf) {
  const int qt = blockIdx.x & 7;
  const int h  = (blockIdx.x >> 3) & 15;
  const int b  = blockIdx.x >> 7;
  __shared__ bf16 qs[128 * 64];
  __shared__ bf16 ks[128 * 64];
  __shared__ bf16 vs[48 * 136];        // transposed: vs[d][t]
  __shared__ bf16 ps[8 * 16 * 136];    // per-wave P tile [16][128]
  const int tid = threadIdx.x, lane = tid & 31, wave = tid >> 5;
  const int ln = lane & 15, hi = lane >> 4;
  const float slope = __expf(log_slopes[h]);
  const float scale = 0.14433756729740643f;  // 1/sqrt(48)
  const size_t baseq = (size_t)b * SEQ * 2304 + (size_t)h * 48;

  for (int i = tid; i < 128 * 64; i += 256) {
    int t = i >> 6, d = i & 63;
    float v = (d < HD) ? qkv[baseq + (size_t)(qt * 128 + t) * 2304 + d] * scale : 0.f;
    qs[t * 64 + d] = (bf16)v;
  }
  float rmax[8], rsum[8];
  v8f oacc[3];
#pragma unroll
  for (int r = 0; r < 8; ++r) { rmax[r] = -3.0e38f; rsum[r] = 0.f; }
#pragma unroll
  for (int t3 = 0; t3 < 3; ++t3) oacc[t3] = (v8f)(0.f);

  for (int kb = 0; kb < 8; ++kb) {
    __syncthreads();
    for (int i = tid; i < 128 * 64; i += 256) {
      int t = i >> 6, d = i & 63;
      float v = (d < HD) ? qkv[baseq + 768 + (size_t)(kb * 128 + t) * 2304 + d] : 0.f;
      ks[t * 64 + d] = (bf16)v;
    }
    for (int i = tid; i < 48 * 128; i += 256) {
      int t = i & 127, d = i >> 7;
      vs[d * 136 + t] = (bf16)qkv[baseq + 1536 + (size_t)(kb * 128 + t) * 2304 + d];
    }
    __syncthreads();

    v16bf aq0 = ld_frag(&qs[(wave * 16 + ln) * 64 + 0], hi);
    v16bf aq1 = ld_frag(&qs[(wave * 16 + ln) * 64 + 32], hi);

    float pv[8][8];
    float bmax[8];
#pragma unroll
    for (int r = 0; r < 8; ++r) bmax[r] = -3.0e38f;
#pragma unroll
    for (int nt = 0; nt < 8; ++nt) {
      v16bf bk0 = ld_frag(&ks[(nt * 16 + ln) * 64 + 0], hi);
      v16bf bk1 = ld_frag(&ks[(nt * 16 + ln) * 64 + 32], hi);
      v8f s = (v8f)(0.f);
      s = wmma_bf(aq0, bk0, s);
      s = wmma_bf(aq1, bk1, s);
      int kg = kb * 128 + nt * 16 + ln;
#pragma unroll
      for (int r = 0; r < 8; ++r) {
        int qg = qt * 128 + wave * 16 + hi * 8 + r;
        float v = s[r] - slope * fabsf((float)(qg - kg));
        pv[nt][r] = v;
        bmax[r] = fmaxf(bmax[r], v);
      }
    }
    // row reductions across the 16 lanes sharing each row (width-16 halves)
#pragma unroll
    for (int r = 0; r < 8; ++r) {
#pragma unroll
      for (int m = 1; m < 16; m <<= 1) bmax[r] = fmaxf(bmax[r], __shfl_xor(bmax[r], m, 16));
    }
    float lsum[8];
#pragma unroll
    for (int r = 0; r < 8; ++r) {
      float mnew = fmaxf(rmax[r], bmax[r]);
      float corr = __expf(rmax[r] - mnew);
      rmax[r] = mnew;
      rsum[r] *= corr;
      lsum[r] = 0.f;
#pragma unroll
      for (int t3 = 0; t3 < 3; ++t3) oacc[t3][r] *= corr;
#pragma unroll
      for (int nt = 0; nt < 8; ++nt) {
        float p = __expf(pv[nt][r] - mnew);
        pv[nt][r] = p;
        lsum[r] += p;
      }
    }
#pragma unroll
    for (int r = 0; r < 8; ++r) {
#pragma unroll
      for (int m = 1; m < 16; m <<= 1) lsum[r] += __shfl_xor(lsum[r], m, 16);
      rsum[r] += lsum[r];
    }
    // scatter P into per-wave LDS tile (row-major [16][128]) for A-layout reload
    bf16* pw = &ps[wave * 16 * 136];
#pragma unroll
    for (int nt = 0; nt < 8; ++nt)
#pragma unroll
      for (int r = 0; r < 8; ++r)
        pw[(hi * 8 + r) * 136 + nt * 16 + ln] = (bf16)pv[nt][r];
    // same-wave LDS ops are in-order; reload as A fragments and multiply by V
#pragma unroll
    for (int kk = 0; kk < 4; ++kk) {
      v16bf ap = ld_frag(&pw[ln * 136 + kk * 32], hi);
#pragma unroll
      for (int t3 = 0; t3 < 3; ++t3) {
        v16bf bv = ld_frag(&vs[(t3 * 16 + ln) * 136 + kk * 32], hi);
        oacc[t3] = wmma_bf(ap, bv, oacc[t3]);
      }
    }
  }
#pragma unroll
  for (int t3 = 0; t3 < 3; ++t3)
#pragma unroll
    for (int r = 0; r < 8; ++r) {
      int t = qt * 128 + wave * 16 + hi * 8 + r;
      int d = h * 48 + t3 * 16 + ln;
      obuf[((size_t)b * SEQ + t) * Dm + d] = (bf16)(oacc[t3][r] / rsum[r]);
    }
}

// ---------------- host launcher ----------------
extern "C" void kernel_launch(void* const* d_in, const int* in_sizes, int n_in,
                              void* d_out, int out_size, void* d_ws, size_t ws_size,
                              hipStream_t stream) {
  const float* cls_tokens = (const float*)d_in[0];
  const float* cls_token  = (const float*)d_in[1];
  const float* log_slopes = (const float*)d_in[2];
  const float* Wqkv = (const float*)d_in[3];
  const float* Wo   = (const float*)d_in[4];
  const float* Wg   = (const float*)d_in[5];
  const float* Wu   = (const float*)d_in[6];
  const float* Wd   = (const float*)d_in[7];
  const float* ln1w = (const float*)d_in[8];
  const float* ln1b = (const float*)d_in[9];
  const float* ln2w = (const float*)d_in[10];
  const float* ln2b = (const float*)d_in[11];
  const float* finw = (const float*)d_in[12];
  const float* finb = (const float*)d_in[13];

  char* ws = (char*)d_ws;
  size_t off = 0;
  auto carve = [&](size_t bytes) { char* p = ws + off; off = (off + bytes + 255) & ~(size_t)255; return p; };

  float* x    = (float*)carve((size_t)Mrow * Dm * 4);
  bf16*  hbuf = (bf16*) carve((size_t)Mrow * Dm * 2);
  float* qkv  = (float*)carve((size_t)Mrow * 3 * Dm * 4);
  bf16*  obuf = (bf16*) carve((size_t)Mrow * Dm * 2);
  float* gate = (float*)carve((size_t)Mrow * FF * 4);
  float* up   = (float*)carve((size_t)Mrow * FF * 4);
  bf16*  act  = (bf16*) carve((size_t)Mrow * FF * 2);
  const size_t nqkv = (size_t)LAY * 3 * Dm * Dm;
  const size_t nwo  = (size_t)LAY * Dm * Dm;
  const size_t nff  = (size_t)LAY * FF * Dm;
  bf16* wqkv_bf = (bf16*)carve(nqkv * 2);
  bf16* wo_bf   = (bf16*)carve(nwo * 2);
  bf16* wg_bf   = (bf16*)carve(nff * 2);
  bf16* wu_bf   = (bf16*)carve(nff * 2);
  bf16* wd_bf   = (bf16*)carve(nff * 2);

  // assemble x and convert weights to bf16
  prep_x<<<(Mrow * Dm) / 256, 256, 0, stream>>>(cls_tokens, cls_token, x);
  f32_to_bf16<<<(unsigned)((nqkv + 255) / 256), 256, 0, stream>>>(Wqkv, wqkv_bf, nqkv);
  f32_to_bf16<<<(unsigned)((nwo  + 255) / 256), 256, 0, stream>>>(Wo, wo_bf, nwo);
  f32_to_bf16<<<(unsigned)((nff  + 255) / 256), 256, 0, stream>>>(Wg, wg_bf, nff);
  f32_to_bf16<<<(unsigned)((nff  + 255) / 256), 256, 0, stream>>>(Wu, wu_bf, nff);
  f32_to_bf16<<<(unsigned)((nff  + 255) / 256), 256, 0, stream>>>(Wd, wd_bf, nff);

  for (int l = 0; l < LAY; ++l) {
    ln_to_bf16<<<Mrow, 256, 0, stream>>>(x, ln1w + l * Dm, ln1b + l * Dm, hbuf);
    gemm_bf16_nt<false><<<dim3((3 * Dm) / 128, Mrow / 128), 256, 0, stream>>>(
        hbuf, wqkv_bf + (size_t)l * 3 * Dm * Dm, nullptr, qkv, Mrow, 3 * Dm, Dm);
    attn_flash<<<Bb * Hh * (SEQ / 128), 256, 0, stream>>>(qkv, log_slopes, obuf);
    gemm_bf16_nt<true><<<dim3(Dm / 128, Mrow / 128), 256, 0, stream>>>(
        obuf, wo_bf + (size_t)l * Dm * Dm, x, x, Mrow, Dm, Dm);
    ln_to_bf16<<<Mrow, 256, 0, stream>>>(x, ln2w + l * Dm, ln2b + l * Dm, hbuf);
    gemm_bf16_nt<false><<<dim3(FF / 128, Mrow / 128), 256, 0, stream>>>(
        hbuf, wg_bf + (size_t)l * FF * Dm, nullptr, gate, Mrow, FF, Dm);
    gemm_bf16_nt<false><<<dim3(FF / 128, Mrow / 128), 256, 0, stream>>>(
        hbuf, wu_bf + (size_t)l * FF * Dm, nullptr, up, Mrow, FF, Dm);
    silu_mul<<<(unsigned)(((size_t)Mrow * FF) / 256), 256, 0, stream>>>(gate, up, act);
    gemm_bf16_nt<true><<<dim3(Dm / 128, Mrow / 128), 256, 0, stream>>>(
        act, wd_bf + (size_t)l * Dm * FF, x, x, Mrow, Dm, FF);
  }
  ln_final<<<Bb, 256, 0, stream>>>(x, finw, finb, (float*)d_out);
}
